// DeepSCTransformerBlock_5531917877947
// MI455X (gfx1250) — compile-verified, matrix operand order checked by reading, not run on hardware
//
#include <hip/hip_runtime.h>
#include <math.h>

// ---------------------------------------------------------------------------
// DeepSC transformer block for MI455X (gfx1250, wave32, WMMA bf16 + TDM path)
// Double-buffered: TDM DMAs tile k+1 while matrix cores consume tile k.
// ---------------------------------------------------------------------------

typedef __attribute__((ext_vector_type(16))) __bf16 v16bf;
typedef __attribute__((ext_vector_type(8)))  float  v8f;
typedef __attribute__((ext_vector_type(4)))  unsigned int v4u;
typedef __attribute__((ext_vector_type(8)))  int v8i;
typedef __attribute__((ext_vector_type(4)))  int v4i;

#define BDIM   2
#define LDIM   2048
#define DDIM   512
#define HHEADS 8
#define HDK    64
#define FFD    2048
#define RROWS  (BDIM * LDIM)

// ---- Tensor Data Mover (async tile DMA into LDS), probe-verified builtin ----
#if defined(__has_builtin)
#if __has_builtin(__builtin_amdgcn_tensor_load_to_lds) && \
    __has_builtin(__builtin_amdgcn_s_wait_tensorcnt)
#define USE_TDM 1
#endif
#endif

#ifdef USE_TDM
#if defined(__clang_major__) && (__clang_major__ >= 23)
#define TDM_CALL(g0, g1, g2, g3) \
  __builtin_amdgcn_tensor_load_to_lds((g0), (g1), (g2), (g3), \
                                      (v8i){0, 0, 0, 0, 0, 0, 0, 0}, 0)
#else
#define TDM_CALL(g0, g1, g2, g3) \
  __builtin_amdgcn_tensor_load_to_lds((g0), (g1), (g2), (g3), 0)
#endif

// 2D tile DMA: tile_x bf16 elements contiguous, tile_y rows, row stride in
// elements; LDS destination padded every (2<<interval_code) DWORDs by
// (amount_code+1) DWORDs -> reproduces our padded LDS strides in hardware.
static __device__ __forceinline__ void tdm_load_2d(unsigned lds_off,
                                                   const void* gsrc,
                                                   unsigned tile_x, unsigned tile_y,
                                                   unsigned row_stride_elems,
                                                   int pad_interval_code,
                                                   int pad_amount_code) {
  unsigned long long ga = (unsigned long long)(size_t)gsrc;
  v4u g0 = { 1u,                                   // count=1, user descriptor
             lds_off,                              // LDS byte address
             (unsigned)(ga & 0xffffffffu),         // global addr lo
             (unsigned)((ga >> 32) & 0x1ffffffu) | (2u << 30) };  // hi + type=2
  int w0 = (1 << 16)                               // data_size = 1 -> 2 bytes
         | (1 << 20)                               // pad_enable
         | (pad_interval_code << 22)
         | (pad_amount_code << 25);
  v8i g1 = { w0,
             (int)(tile_x << 16),                  // tensor_dim0 lo16 @ [31:16]
             (int)(tile_y << 16),                  // tensor_dim1 lo16 @ [31:16]
             (int)(tile_x << 16),                  // tile_dim0 @ [31:16]
             (int)tile_y,                          // tile_dim1 @ [15:0]
             (int)row_stride_elems,                // tensor_dim0_stride lo32
             0, 0 };
  v4i z4 = {0, 0, 0, 0};
  TDM_CALL(g0, g1, z4, z4);
}
#endif  // USE_TDM

static __device__ __forceinline__ v8f wmma_bf16(v16bf a, v16bf b, v8f c) {
  return __builtin_amdgcn_wmma_f32_16x16x32_bf16(false, a, false, b,
                                                 (short)0, c, false, false);
}

// CDNA5 16-bit fragment: per-lane two contiguous 8-elem chunks at +0 / +16.
static __device__ __forceinline__ void ld_frag(v16bf& f, const __bf16* p) {
  ((uint4*)&f)[0] = *(const uint4*)(p);
  ((uint4*)&f)[1] = *(const uint4*)(p + 16);
}

// ---------------------------------------------------------------------------
// f32 -> bf16 conversion
// ---------------------------------------------------------------------------
__global__ void cvt_bf16_kernel(const float* __restrict__ s,
                                __bf16* __restrict__ d, int n) {
  int i = (blockIdx.x * blockDim.x + threadIdx.x) * 4;
  if (i + 3 < n) {
    float4 f = *(const float4*)&s[i];
    d[i + 0] = (__bf16)f.x; d[i + 1] = (__bf16)f.y;
    d[i + 2] = (__bf16)f.z; d[i + 3] = (__bf16)f.w;
  } else {
    for (int j = i; j < n; ++j) d[j] = (__bf16)s[j];
  }
}

__global__ void concat_bf16_kernel(const float* __restrict__ g,
                                   const float* __restrict__ e,
                                   __bf16* __restrict__ c) {
  int idx = blockIdx.x * blockDim.x + threadIdx.x;        // over R * 2D
  int row = idx >> 10;                                    // 2D = 1024
  int col = idx & 1023;
  float v = (col < DDIM) ? g[(size_t)row * DDIM + col]
                         : e[(size_t)row * DDIM + (col - DDIM)];
  c[idx] = (__bf16)v;
}

// ---------------------------------------------------------------------------
// WMMA GEMM:  C[M,N] = A[M,K] @ W[N,K]^T + bias  (NT, bf16 in, f32 acc)
// mode 0: bf16 out   mode 1: f32 out + residual   mode 2: GELU -> bf16 out
// Block 256 (8 waves), tile 128x128, K-step 32; double-buffered TDM staging.
// ---------------------------------------------------------------------------
#define GEMM_LDSS 40   // padded bf16 stride: 32 elems + 4-DWORD TDM pad = 80B

__global__ __launch_bounds__(256)
void gemm_wmma_kernel(const __bf16* __restrict__ A, const __bf16* __restrict__ W,
                      const float* __restrict__ bias, const float* __restrict__ resid,
                      void* __restrict__ out, int N, int K, int mode) {
#ifdef USE_TDM
  __shared__ __align__(16) __bf16 As[2][128 * GEMM_LDSS];
  __shared__ __align__(16) __bf16 Bs[2][128 * GEMM_LDSS];
#else
  __shared__ __align__(16) __bf16 As[1][128 * GEMM_LDSS];
  __shared__ __align__(16) __bf16 Bs[1][128 * GEMM_LDSS];
#endif

  const int tid  = threadIdx.x;
  const int lane = tid & 31;
  const int w    = tid >> 5;
  const int wr   = w >> 2;           // 0..1  (64 rows each)
  const int wc   = w & 3;            // 0..3  (32 cols each)
  const int l15  = lane & 15;
  const int hl   = lane >> 4;
  const int m0   = blockIdx.y * 128;
  const int n0   = blockIdx.x * 128;

  v8f acc[4][2];
#pragma unroll
  for (int mt = 0; mt < 4; ++mt)
#pragma unroll
    for (int nt = 0; nt < 2; ++nt)
#pragma unroll
      for (int i = 0; i < 8; ++i) acc[mt][nt][i] = 0.f;

  int ib = 0;
#ifdef USE_TDM
  const unsigned asOff[2] = { (unsigned)(size_t)&As[0][0],
                              (unsigned)(size_t)&As[1][0] };
  const unsigned bsOff[2] = { (unsigned)(size_t)&Bs[0][0],
                              (unsigned)(size_t)&Bs[1][0] };
  if (tid < 32) {                    // prologue: DMA first k-tile into buf 0
    tdm_load_2d(asOff[0], &A[(size_t)m0 * K], 32u, 128u, (unsigned)K, 3, 3);
    tdm_load_2d(bsOff[0], &W[(size_t)n0 * K], 32u, 128u, (unsigned)K, 3, 3);
  }
#endif

  for (int kk = 0; kk < K; kk += 32) {
#ifdef USE_TDM
    if (tid < 32) __builtin_amdgcn_s_wait_tensorcnt(0);   // buf ib complete
    __syncthreads();                                      // visible; ib^1 free
    if (kk + 32 < K && tid < 32) {   // overlap: DMA next tile during compute
      tdm_load_2d(asOff[ib ^ 1], &A[(size_t)m0 * K + kk + 32],
                  32u, 128u, (unsigned)K, 3, 3);
      tdm_load_2d(bsOff[ib ^ 1], &W[(size_t)n0 * K + kk + 32],
                  32u, 128u, (unsigned)K, 3, 3);
    }
#else
    __syncthreads();
    uint4 areg[2], breg[2];
#pragma unroll
    for (int i = 0; i < 2; ++i) {
      int c  = tid + i * 256;
      int r  = c >> 2;
      int cc = (c & 3) << 3;
      areg[i] = *(const uint4*)&A[(size_t)(m0 + r) * K + kk + cc];
      breg[i] = *(const uint4*)&W[(size_t)(n0 + r) * K + kk + cc];
    }
#pragma unroll
    for (int i = 0; i < 2; ++i) {
      int c  = tid + i * 256;
      int r  = c >> 2;
      int cc = (c & 3) << 3;
      *(uint4*)&As[0][r * GEMM_LDSS + cc] = areg[i];
      *(uint4*)&Bs[0][r * GEMM_LDSS + cc] = breg[i];
    }
    __syncthreads();
#endif

    v16bf afr[4], bfr[2];
#pragma unroll
    for (int mt = 0; mt < 4; ++mt) {
      int row = wr * 64 + mt * 16 + l15;
      ld_frag(afr[mt], &As[ib][row * GEMM_LDSS + hl * 8]);
    }
#pragma unroll
    for (int nt = 0; nt < 2; ++nt) {
      int row = wc * 32 + nt * 16 + l15;
      ld_frag(bfr[nt], &Bs[ib][row * GEMM_LDSS + hl * 8]);
    }
#pragma unroll
    for (int mt = 0; mt < 4; ++mt)
#pragma unroll
      for (int nt = 0; nt < 2; ++nt)
        acc[mt][nt] = wmma_bf16(afr[mt], bfr[nt], acc[mt][nt]);
#ifdef USE_TDM
    ib ^= 1;
#endif
  }

  // epilogue (C layout: row = i + 8*hl, col = l15)
#pragma unroll
  for (int mt = 0; mt < 4; ++mt)
#pragma unroll
    for (int nt = 0; nt < 2; ++nt) {
      int gcol = n0 + wc * 32 + nt * 16 + l15;
      float bv = bias ? bias[gcol] : 0.f;
#pragma unroll
      for (int i = 0; i < 8; ++i) {
        int grow = m0 + wr * 64 + mt * 16 + i + 8 * hl;
        size_t o = (size_t)grow * N + gcol;
        float v  = acc[mt][nt][i] + bv;
        if (mode == 0) {
          ((__bf16*)out)[o] = (__bf16)v;
        } else if (mode == 1) {
          ((float*)out)[o] = v + (resid ? resid[o] : 0.f);
        } else {
          float g = 0.5f * v * (1.f + erff(v * 0.70710678118f));
          ((__bf16*)out)[o] = (__bf16)g;
        }
      }
    }
}

// ---------------------------------------------------------------------------
// Flash attention with post-softmax mask + L1 renormalization.
//   A_bar_j = e_j*M_j / (S + eps*Z),  e_j = exp(s_j - m), Z = sum e, S = sum e*M
// Grid (L/64, H, B), block 128 (4 waves x 16 query rows).
// K tile: double-buffered TDM DMA. V tile: manual transpose, regs prefetched.
// ---------------------------------------------------------------------------
#define FA_LDSS 72   // padded bf16 stride: 64 elems + 4-DWORD TDM pad = 144B

__global__ __launch_bounds__(128)
void flash_attn_kernel(const __bf16* __restrict__ Q, const __bf16* __restrict__ Kb,
                       const __bf16* __restrict__ Vb, const float* __restrict__ Mask,
                       __bf16* __restrict__ Octx) {
#ifdef USE_TDM
  __shared__ __align__(16) __bf16 kT[2][64 * FA_LDSS];    // [key][hd]
#else
  __shared__ __align__(16) __bf16 kT[1][64 * FA_LDSS];
#endif
  __shared__ __align__(16) __bf16 vT[64 * FA_LDSS];       // [hd][key] (transposed)
  __shared__ __align__(16) __bf16 pS[4][16 * FA_LDSS];    // per-wave P staging

  const int b = blockIdx.z, h = blockIdx.y;
  const int q0 = blockIdx.x * 64;
  const int tid = threadIdx.x, lane = tid & 31, w = tid >> 5;
  const int l15 = lane & 15, hl = lane >> 4;
  const int qw = q0 + w * 16;

  // Q fragments held in registers across all key blocks
  v16bf qf[2];
  {
    const __bf16* qp = &Q[((size_t)(b * LDIM + qw + l15)) * DDIM + h * HDK];
#pragma unroll
    for (int c = 0; c < 2; ++c) ld_frag(qf[c], qp + c * 32 + hl * 8);
  }

  float mo[8], zz[8], ss[8];
  v8f oacc[4];
#pragma unroll
  for (int i = 0; i < 8; ++i) { mo[i] = -1e30f; zz[i] = 0.f; ss[i] = 0.f; }
#pragma unroll
  for (int nt = 0; nt < 4; ++nt)
#pragma unroll
    for (int i = 0; i < 8; ++i) oacc[nt][i] = 0.f;

  int ib = 0;
  uint4 vreg[4];
#ifdef USE_TDM
  const unsigned ktOff[2] = { (unsigned)(size_t)&kT[0][0],
                              (unsigned)(size_t)&kT[1][0] };
  if (tid < 32)                      // prologue: DMA K tile of block 0
    tdm_load_2d(ktOff[0], &Kb[((size_t)(b * LDIM)) * DDIM + h * HDK],
                64u, 64u, (unsigned)DDIM, 4, 3);
#endif
  // prologue: V regs for block 0
#pragma unroll
  for (int i = 0; i < 4; ++i) {
    int c  = tid + i * 128;
    int r  = c >> 3;
    int cc = (c & 7) << 3;
    vreg[i] = *(const uint4*)&Vb[((size_t)(b * LDIM + r)) * DDIM + h * HDK + cc];
  }

  for (int kb = 0; kb < LDIM / 64; ++kb) {
    __syncthreads();                 // prior block's vT/pS reads complete
    // scatter prefetched V regs into transposed vT
#pragma unroll
    for (int i = 0; i < 4; ++i) {
      int c  = tid + i * 128;
      int r  = c >> 3;
      int cc = (c & 7) << 3;
#ifndef USE_TDM
      *(uint4*)&kT[0][r * FA_LDSS + cc] =
          *(const uint4*)&Kb[((size_t)(b * LDIM + kb * 64 + r)) * DDIM + h * HDK + cc];
#endif
      const __bf16* ve = (const __bf16*)&vreg[i];
#pragma unroll
      for (int j = 0; j < 8; ++j) vT[(cc + j) * FA_LDSS + r] = ve[j];
    }
#ifdef USE_TDM
    if (tid < 32) __builtin_amdgcn_s_wait_tensorcnt(0);   // kT[ib] complete
#endif
    __syncthreads();                 // kT[ib] + vT visible to all waves
    if (kb + 1 < LDIM / 64) {
#ifdef USE_TDM
      if (tid < 32)                  // overlap: DMA next K tile during compute
        tdm_load_2d(ktOff[ib ^ 1],
                    &Kb[((size_t)(b * LDIM + (kb + 1) * 64)) * DDIM + h * HDK],
                    64u, 64u, (unsigned)DDIM, 4, 3);
#endif
      // overlap: prefetch next V tile into registers
#pragma unroll
      for (int i = 0; i < 4; ++i) {
        int c  = tid + i * 128;
        int r  = c >> 3;
        int cc = (c & 7) << 3;
        vreg[i] = *(const uint4*)
            &Vb[((size_t)(b * LDIM + (kb + 1) * 64 + r)) * DDIM + h * HDK + cc];
      }
    }

    // S = (Q @ K^T) * 1/sqrt(HD)
    v8f sacc[4];
#pragma unroll
    for (int t = 0; t < 4; ++t)
#pragma unroll
      for (int i = 0; i < 8; ++i) sacc[t][i] = 0.f;
#pragma unroll
    for (int t = 0; t < 4; ++t)
#pragma unroll
      for (int c = 0; c < 2; ++c) {
        v16bf kf;
        ld_frag(kf, &kT[ib][(t * 16 + l15) * FA_LDSS + c * 32 + hl * 8]);
        sacc[t] = wmma_bf16(qf[c], kf, sacc[t]);
      }

    // online softmax with mask + dual running sums
    float rm[8], mn[8], rs[8], zp[8], sp[8];
#pragma unroll
    for (int i = 0; i < 8; ++i) rm[i] = -1e30f;
#pragma unroll
    for (int t = 0; t < 4; ++t)
#pragma unroll
      for (int i = 0; i < 8; ++i) {
        sacc[t][i] *= 0.125f;        // 1/sqrt(64)
        rm[i] = fmaxf(rm[i], sacc[t][i]);
      }
#pragma unroll
    for (int i = 0; i < 8; ++i) {
#pragma unroll
      for (int off = 1; off < 16; off <<= 1)
        rm[i] = fmaxf(rm[i], __shfl_xor(rm[i], off, 32));
      mn[i] = fmaxf(mo[i], rm[i]);
      rs[i] = __expf(mo[i] - mn[i]);
      zp[i] = 0.f; sp[i] = 0.f;
    }
#pragma unroll
    for (int t = 0; t < 4; ++t) {
      int col = kb * 64 + t * 16 + l15;
#pragma unroll
      for (int i = 0; i < 8; ++i) {
        int row  = qw + i + 8 * hl;
        float e  = __expf(sacc[t][i] - mn[i]);
        float pm = e * Mask[((size_t)(b * LDIM + row)) * LDIM + col];
        zp[i] += e;
        sp[i] += pm;
        pS[w][(i + 8 * hl) * FA_LDSS + t * 16 + l15] = (__bf16)pm;
      }
    }
#pragma unroll
    for (int i = 0; i < 8; ++i) {
#pragma unroll
      for (int off = 1; off < 16; off <<= 1) {
        zp[i] += __shfl_xor(zp[i], off, 32);
        sp[i] += __shfl_xor(sp[i], off, 32);
      }
      zz[i] = zz[i] * rs[i] + zp[i];
      ss[i] = ss[i] * rs[i] + sp[i];
      mo[i] = mn[i];
    }
#pragma unroll
    for (int nt = 0; nt < 4; ++nt)
#pragma unroll
      for (int i = 0; i < 8; ++i) oacc[nt][i] *= rs[i];

    // O += P @ V   (A = P from wave-private LDS, B = V^T columns)
#pragma unroll
    for (int c = 0; c < 2; ++c) {
      v16bf pf;
      ld_frag(pf, &pS[w][l15 * FA_LDSS + c * 32 + hl * 8]);
#pragma unroll
      for (int nt = 0; nt < 4; ++nt) {
        v16bf vf;
        ld_frag(vf, &vT[(nt * 16 + l15) * FA_LDSS + c * 32 + hl * 8]);
        oacc[nt] = wmma_bf16(pf, vf, oacc[nt]);
      }
    }
#ifdef USE_TDM
    ib ^= 1;
#endif
  }

  float inv[8];
#pragma unroll
  for (int i = 0; i < 8; ++i) inv[i] = 1.f / (ss[i] + 1e-8f * zz[i]);
#pragma unroll
  for (int nt = 0; nt < 4; ++nt) {
    int col = h * HDK + nt * 16 + l15;
#pragma unroll
    for (int i = 0; i < 8; ++i) {
      int row = qw + i + 8 * hl;
      Octx[((size_t)(b * LDIM + row)) * DDIM + col] = (__bf16)(oacc[nt][i] * inv[i]);
    }
  }
}

// ---------------------------------------------------------------------------
// LayerNorm over D=512; one wave per row; writes f32 and/or bf16.
// ---------------------------------------------------------------------------
__global__ __launch_bounds__(256)
void layernorm_kernel(const float* __restrict__ X, const float* __restrict__ gamma,
                      const float* __restrict__ beta, float* __restrict__ Yf,
                      __bf16* __restrict__ Ybf) {
  const int w = threadIdx.x >> 5, lane = threadIdx.x & 31;
  const int row = blockIdx.x * 8 + w;
  const float* x = &X[(size_t)row * DDIM];

  float v[16];
  float sum = 0.f;
#pragma unroll
  for (int j = 0; j < 16; ++j) { v[j] = x[lane + j * 32]; sum += v[j]; }
#pragma unroll
  for (int off = 16; off; off >>= 1) sum += __shfl_xor(sum, off, 32);
  float mu = sum * (1.f / 512.f);
  float vs = 0.f;
#pragma unroll
  for (int j = 0; j < 16; ++j) { float d = v[j] - mu; vs += d * d; }
#pragma unroll
  for (int off = 16; off; off >>= 1) vs += __shfl_xor(vs, off, 32);
  float rstd = rsqrtf(vs * (1.f / 512.f) + 1e-5f);
#pragma unroll
  for (int j = 0; j < 16; ++j) {
    int c = lane + j * 32;
    float y = (v[j] - mu) * rstd * gamma[c] + beta[c];
    if (Yf)  Yf[(size_t)row * DDIM + c] = y;
    if (Ybf) Ybf[(size_t)row * DDIM + c] = (__bf16)y;
  }
}

// ---------------------------------------------------------------------------
// Host orchestration
// ---------------------------------------------------------------------------
extern "C" void kernel_launch(void* const* d_in, const int* in_sizes, int n_in,
                              void* d_out, int out_size, void* d_ws, size_t ws_size,
                              hipStream_t stream) {
  (void)in_sizes; (void)n_in; (void)out_size; (void)ws_size;

  const float* gene    = (const float*)d_in[0];
  const float* expr    = (const float*)d_in[1];
  const float* Mask    = (const float*)d_in[2];
  const float* Wv_gene = (const float*)d_in[3];
  const float* Wv_expr = (const float*)d_in[4];
  const float* Wq_g    = (const float*)d_in[5];
  const float* Wk_g    = (const float*)d_in[6];
  const float* Wo_g    = (const float*)d_in[7];
  const float* Wq_e    = (const float*)d_in[8];
  const float* Wk_e    = (const float*)d_in[9];
  const float* Wo_e    = (const float*)d_in[10];
  const float* bv_gene = (const float*)d_in[11];
  const float* bv_expr = (const float*)d_in[12];
  const float* bq_g    = (const float*)d_in[13];
  const float* bk_g    = (const float*)d_in[14];
  const float* bo_g    = (const float*)d_in[15];
  const float* bq_e    = (const float*)d_in[16];
  const float* bk_e    = (const float*)d_in[17];
  const float* bo_e    = (const float*)d_in[18];
  const float* bf_     = (const float*)d_in[19];
  const float* b2_g    = (const float*)d_in[20];
  const float* b2_e    = (const float*)d_in[21];
  const float* Wf      = (const float*)d_in[22];
  const float* W1_g    = (const float*)d_in[23];
  const float* b1_g    = (const float*)d_in[24];
  const float* W2_g    = (const float*)d_in[25];
  const float* W1_e    = (const float*)d_in[26];
  const float* b1_e    = (const float*)d_in[27];
  const float* W2_e    = (const float*)d_in[28];
  const float* gamma_g1 = (const float*)d_in[29];
  const float* gamma_g2 = (const float*)d_in[30];
  const float* gamma_e1 = (const float*)d_in[31];
  const float* gamma_e2 = (const float*)d_in[32];
  const float* beta_g1  = (const float*)d_in[33];
  const float* beta_g2  = (const float*)d_in[34];
  const float* beta_e1  = (const float*)d_in[35];
  const float* beta_e2  = (const float*)d_in[36];

  char* wsb = (char*)d_ws;
  size_t bump = 0;
  auto alloc = [&](size_t bytes) -> void* {
    void* p = wsb + bump;
    bump += (bytes + 255) & ~(size_t)255;
    return p;
  };

  const size_t RD  = (size_t)RROWS * DDIM;
  const size_t DD  = (size_t)DDIM * DDIM;
  const size_t RFF = (size_t)RROWS * FFD;

  __bf16* gene_bf = (__bf16*)alloc(RD * 2);
  __bf16* expr_bf = (__bf16*)alloc(RD * 2);
  __bf16* cat_bf  = (__bf16*)alloc(RD * 4);
  __bf16* wvg_bf  = (__bf16*)alloc(DD * 2);
  __bf16* wve_bf  = (__bf16*)alloc(DD * 2);
  __bf16* wqg_bf  = (__bf16*)alloc(DD * 2);
  __bf16* wkg_bf  = (__bf16*)alloc(DD * 2);
  __bf16* wog_bf  = (__bf16*)alloc(DD * 2);
  __bf16* wqe_bf  = (__bf16*)alloc(DD * 2);
  __bf16* wke_bf  = (__bf16*)alloc(DD * 2);
  __bf16* woe_bf  = (__bf16*)alloc(DD * 2);
  __bf16* wf_bf   = (__bf16*)alloc(DD * 2 * 2);
  __bf16* w1g_bf  = (__bf16*)alloc((size_t)FFD * DDIM * 2);
  __bf16* w2g_bf  = (__bf16*)alloc((size_t)DDIM * FFD * 2);
  __bf16* w1e_bf  = (__bf16*)alloc((size_t)FFD * DDIM * 2);
  __bf16* w2e_bf  = (__bf16*)alloc((size_t)DDIM * FFD * 2);
  __bf16* qg_bf   = (__bf16*)alloc(RD * 2);
  __bf16* kg_bf   = (__bf16*)alloc(RD * 2);
  __bf16* vg_bf   = (__bf16*)alloc(RD * 2);
  __bf16* qe_bf   = (__bf16*)alloc(RD * 2);
  __bf16* ke_bf   = (__bf16*)alloc(RD * 2);
  __bf16* ve_bf   = (__bf16*)alloc(RD * 2);
  __bf16* fused_bf = (__bf16*)alloc(RD * 2);
  __bf16* ctxg_bf = (__bf16*)alloc(RD * 2);
  __bf16* ctxe_bf = (__bf16*)alloc(RD * 2);
  float*  attng_f = (float*)alloc(RD * 4);
  float*  attne_f = (float*)alloc(RD * 4);
  float*  hg_f    = (float*)alloc(RD * 4);
  float*  he_f    = (float*)alloc(RD * 4);
  __bf16* hg_bf   = (__bf16*)alloc(RD * 2);
  __bf16* he_bf   = (__bf16*)alloc(RD * 2);
  __bf16* ffn1g_bf = (__bf16*)alloc(RFF * 2);
  __bf16* ffn1e_bf = (__bf16*)alloc(RFF * 2);
  float*  ffn2g_f = (float*)alloc(RD * 4);
  float*  ffn2e_f = (float*)alloc(RD * 4);

  auto cvt = [&](const float* s, __bf16* d, size_t n) {
    int blocks = (int)((n / 4 + 255) / 256);
    cvt_bf16_kernel<<<blocks, 256, 0, stream>>>(s, d, (int)n);
  };
  auto gemm = [&](const __bf16* A, const __bf16* W, const float* bias,
                  const float* resid, void* out, int N, int K, int mode) {
    dim3 g(N / 128, RROWS / 128);
    gemm_wmma_kernel<<<g, 256, 0, stream>>>(A, W, bias, resid, out, N, K, mode);
  };
  auto ln = [&](const float* x, const float* g, const float* be,
                float* yf, __bf16* ybf) {
    layernorm_kernel<<<RROWS / 8, 256, 0, stream>>>(x, g, be, yf, ybf);
  };

  // ---- convert activations & weights to bf16 ----
  cvt(gene, gene_bf, RD);
  cvt(expr, expr_bf, RD);
  concat_bf16_kernel<<<(int)(RD * 2 / 256), 256, 0, stream>>>(gene, expr, cat_bf);
  cvt(Wv_gene, wvg_bf, DD);  cvt(Wv_expr, wve_bf, DD);
  cvt(Wq_g, wqg_bf, DD);     cvt(Wk_g, wkg_bf, DD);    cvt(Wo_g, wog_bf, DD);
  cvt(Wq_e, wqe_bf, DD);     cvt(Wk_e, wke_bf, DD);    cvt(Wo_e, woe_bf, DD);
  cvt(Wf, wf_bf, DD * 2);
  cvt(W1_g, w1g_bf, (size_t)FFD * DDIM);  cvt(W2_g, w2g_bf, (size_t)DDIM * FFD);
  cvt(W1_e, w1e_bf, (size_t)FFD * DDIM);  cvt(W2_e, w2e_bf, (size_t)DDIM * FFD);

  // ---- projections (bf16 out) ----
  gemm(gene_bf, wvg_bf, bv_gene, nullptr, vg_bf, DDIM, DDIM, 0);
  gemm(gene_bf, wqg_bf, bq_g,   nullptr, qg_bf, DDIM, DDIM, 0);
  gemm(gene_bf, wkg_bf, bk_g,   nullptr, kg_bf, DDIM, DDIM, 0);
  gemm(cat_bf,  wf_bf,  bf_,    nullptr, fused_bf, DDIM, 2 * DDIM, 0);
  gemm(expr_bf, wve_bf, bv_expr, nullptr, ve_bf, DDIM, DDIM, 0);
  gemm(fused_bf, wqe_bf, bq_e,  nullptr, qe_bf, DDIM, DDIM, 0);
  gemm(fused_bf, wke_bf, bk_e,  nullptr, ke_bf, DDIM, DDIM, 0);

  // ---- flash attention (masked + L1 renorm) ----
  dim3 fg(LDIM / 64, HHEADS, BDIM);
  flash_attn_kernel<<<fg, 128, 0, stream>>>(qg_bf, kg_bf, vg_bf, Mask, ctxg_bf);
  flash_attn_kernel<<<fg, 128, 0, stream>>>(qe_bf, ke_bf, ve_bf, Mask, ctxe_bf);

  // ---- output projection + residual (f32 out) ----
  gemm(ctxg_bf, wog_bf, bo_g, gene, attng_f, DDIM, DDIM, 1);
  gemm(ctxe_bf, woe_bf, bo_e, expr, attne_f, DDIM, DDIM, 1);

  // ---- LN1 ----
  ln(attng_f, gamma_g1, beta_g1, hg_f, hg_bf);
  ln(attne_f, gamma_e1, beta_e1, he_f, he_bf);

  // ---- FFN ----
  gemm(hg_bf, w1g_bf, b1_g, nullptr, ffn1g_bf, FFD, DDIM, 2);      // GELU
  gemm(ffn1g_bf, w2g_bf, b2_g, hg_f, ffn2g_f, DDIM, FFD, 1);
  gemm(he_bf, w1e_bf, b1_e, nullptr, ffn1e_bf, FFD, DDIM, 2);
  gemm(ffn1e_bf, w2e_bf, b2_e, he_f, ffn2e_f, DDIM, FFD, 1);

  // ---- LN2 -> outputs ----
  float* out_gene = (float*)d_out;
  float* out_expr = out_gene + RD;
  ln(ffn2g_f, gamma_g2, beta_g2, out_gene, nullptr);
  ln(ffn2e_f, gamma_e2, beta_e2, out_expr, nullptr);
}
